// CustomAttention_60129542363
// MI455X (gfx1250) — matrix operand order repaired and checked by reading, not compiled, not run on hardware
//
#include <hip/hip_runtime.h>

#define BATCH 4
#define HEADS 16
#define SEQ   2048
#define DHEAD 64
#define TQ    256          // query rows per workgroup (8 waves x 32 rows)
#define CH    64           // key/value chunk columns
#define NW    8
#define NCH   (SEQ / CH)
#define QSTR  72           // Q/W staging row stride (bf16), 144B = 9*16 -> 16B aligned rows
#define KSTR  72
#define VPSTR 136          // V pair-interleaved row stride (bf16), 272B = 17*16

typedef __attribute__((ext_vector_type(16))) __bf16 v16bf;
typedef __attribute__((ext_vector_type(8)))  float  v8f;

#define EXP2F(x) __builtin_amdgcn_exp2f(x)   // single v_exp_f32 (hardware exp is base-2)

__device__ __forceinline__ float redmax16(float v) {
  #pragma unroll
  for (int o = 1; o < 16; o <<= 1) v = fmaxf(v, __shfl_xor(v, o, 32));
  return v;
}
__device__ __forceinline__ float redsum16(float v) {
  #pragma unroll
  for (int o = 1; o < 16; o <<= 1) v += __shfl_xor(v, o, 32);
  return v;
}

// 8 f32 -> 8 bf16, one 16B LDS store
__device__ __forceinline__ void pack8(float4 a, float4 b, __bf16* dst) {
  __bf16 t[8] = {(__bf16)a.x, (__bf16)a.y, (__bf16)a.z, (__bf16)a.w,
                 (__bf16)b.x, (__bf16)b.y, (__bf16)b.z, (__bf16)b.w};
  *reinterpret_cast<uint4*>(dst) = *reinterpret_cast<const uint4*>(t);
}
// row-pair interleave: dst[2e] = rowEven[e], dst[2e+1] = rowOdd[e]
__device__ __forceinline__ void pack8i(float4 a, float4 b, __bf16* dst) {
  __bf16 t[8] = {(__bf16)a.x, (__bf16)b.x, (__bf16)a.y, (__bf16)b.y,
                 (__bf16)a.z, (__bf16)b.z, (__bf16)a.w, (__bf16)b.w};
  *reinterpret_cast<uint4*>(dst) = *reinterpret_cast<const uint4*>(t);
}

__global__ __launch_bounds__(256) void attn_fwd(
    const float* __restrict__ Q, const float* __restrict__ K,
    const float* __restrict__ V, float* __restrict__ O,
    float* __restrict__ W)
{
  // QW: Q tile during pass 1 fragment build, then per-wave W staging in pass 2
  // (same shape: row (wid*32 + r) x QSTR)
  __shared__ __align__(16) __bf16 QW[TQ * QSTR];
  __shared__ __align__(16) __bf16 Ks[CH * KSTR];
  __shared__ __align__(16) __bf16 Vp[(CH / 2) * VPSTR];

  const int tid  = threadIdx.x;
  const int wid  = tid >> 5;
  const int lane = tid & 31;
  const int h    = lane >> 4;
  const int m    = lane & 15;

  const int bh = blockIdx.y;
  const int qt = blockIdx.x;

  const float* Qg = Q + ((size_t)bh * SEQ + (size_t)qt * TQ) * DHEAD;
  const float* Kg = K + (size_t)bh * SEQ * DHEAD;
  const float* Vg = V + (size_t)bh * SEQ * DHEAD;
  float*       Og = O + ((size_t)bh * SEQ + (size_t)qt * TQ) * DHEAD;
  float*       Wg = W + ((size_t)bh * SEQ + (size_t)qt * TQ) * SEQ;

  // ---------------- Q tile: f32 global -> bf16 LDS (16B stores) ----------------
  #pragma unroll
  for (int t = tid; t < TQ * 8; t += 256) {      // (row, group-of-8-cols)
    int row = t >> 3, cg = t & 7;
    const float4* s = (const float4*)(Qg + (size_t)row * DHEAD + cg * 8);
    pack8(s[0], s[1], &QW[row * QSTR + cg * 8]);
  }
  __syncthreads();

  // Persistent Q A-fragments: 2 row-tiles x 2 k-steps (contiguous k = 16h + s per lane half)
  v16bf aQ[2][2];
  #pragma unroll
  for (int t2 = 0; t2 < 2; ++t2) {
    const __bf16* p = &QW[(wid * 32 + t2 * 16 + m) * QSTR];
    #pragma unroll
    for (int j = 0; j < 8; ++j) {
      int kk = 16 * h + 2 * j;
      aQ[t2][0][2 * j]     = p[kk];
      aQ[t2][0][2 * j + 1] = p[kk + 1];
      aQ[t2][1][2 * j]     = p[kk + 32];
      aQ[t2][1][2 * j + 1] = p[kk + 33];
    }
  }

  const float scl = 0.125f * 1.44269504088896f;   // log2(e)/sqrt(D): exp2 domain
  float Ml[16], Sl[16];                           // per-LANE online max/sum
  #pragma unroll
  for (int i = 0; i < 16; ++i) { Ml[i] = -3.0e38f; Sl[i] = 0.0f; }

  // ================= pass 1: per-lane online max + sumexp (no writes) =================
  for (int c = 0; c < NCH; ++c) {
    __syncthreads();
    #pragma unroll
    for (int t = tid; t < CH * 8; t += 256) {
      int row = t >> 3, cg = t & 7;
      const float4* s = (const float4*)(Kg + ((size_t)c * CH + row) * DHEAD + cg * 8);
      pack8(s[0], s[1], &Ks[row * KSTR + cg * 8]);
    }
    if (c + 1 < NCH)
      __builtin_prefetch(Kg + ((size_t)(c + 1) * CH) * DHEAD + tid * 16, 0, 0);
    __syncthreads();

    #pragma unroll
    for (int nt = 0; nt < 4; ++nt) {
      v16bf b0, b1;
      {
        const __bf16* p = &Ks[(nt * 16 + m) * KSTR];
        #pragma unroll
        for (int j = 0; j < 8; ++j) {
          int kk = 16 * h + 2 * j;
          b0[2 * j]     = p[kk];      b0[2 * j + 1] = p[kk + 1];
          b1[2 * j]     = p[kk + 32]; b1[2 * j + 1] = p[kk + 33];
        }
      }
      v8f acc[2] = {};
      acc[0] = __builtin_amdgcn_wmma_f32_16x16x32_bf16(false, aQ[0][0], false, b0, (short)0, acc[0], false, false);
      acc[0] = __builtin_amdgcn_wmma_f32_16x16x32_bf16(false, aQ[0][1], false, b1, (short)0, acc[0], false, false);
      acc[1] = __builtin_amdgcn_wmma_f32_16x16x32_bf16(false, aQ[1][0], false, b0, (short)0, acc[1], false, false);
      acc[1] = __builtin_amdgcn_wmma_f32_16x16x32_bf16(false, aQ[1][1], false, b1, (short)0, acc[1], false, false);
      #pragma unroll
      for (int t2 = 0; t2 < 2; ++t2) {
        #pragma unroll
        for (int r = 0; r < 8; ++r) {
          int i = t2 * 8 + r;
          float sc = acc[t2][r] * scl;
          float nm = fmaxf(Ml[i], sc);
          Sl[i] = Sl[i] * EXP2F(Ml[i] - nm) + EXP2F(sc - nm);
          Ml[i] = nm;
        }
      }
    }
  }

  // one cross-lane merge at the end (128 bpermutes/wave instead of ~16K)
  float Mrow[16], invS[16];
  #pragma unroll
  for (int i = 0; i < 16; ++i) {
    float M = redmax16(Ml[i]);
    float S = redsum16(Sl[i] * EXP2F(Ml[i] - M));
    Mrow[i] = M;
    invS[i] = 1.0f / S;
  }

  v8f oacc[2][4];
  #pragma unroll
  for (int t2 = 0; t2 < 2; ++t2)
    #pragma unroll
    for (int dt = 0; dt < 4; ++dt) oacc[t2][dt] = (v8f){};

  // ====== pass 2: recompute scores, write weights once, fused PV GEMM ======
  for (int c = 0; c < NCH; ++c) {
    __syncthreads();
    #pragma unroll
    for (int t = tid; t < CH * 8; t += 256) {
      int row = t >> 3, cg = t & 7;
      const float4* s = (const float4*)(Kg + ((size_t)c * CH + row) * DHEAD + cg * 8);
      pack8(s[0], s[1], &Ks[row * KSTR + cg * 8]);
    }
    #pragma unroll
    for (int t = tid; t < (CH / 2) * 16; t += 256) {   // (row-pair, group-of-4-cols)
      int p = t >> 4, cg = t & 15;
      const float* base = Vg + ((size_t)c * CH + 2 * p) * DHEAD + cg * 4;
      float4 a = *(const float4*)base;
      float4 b = *(const float4*)(base + DHEAD);
      pack8i(a, b, &Vp[p * VPSTR + cg * 8]);
    }
    if (c + 1 < NCH) {
      __builtin_prefetch(Kg + ((size_t)(c + 1) * CH) * DHEAD + tid * 16, 0, 0);
      __builtin_prefetch(Vg + ((size_t)(c + 1) * CH) * DHEAD + tid * 16, 0, 0);
    }
    __syncthreads();

    #pragma unroll
    for (int nt = 0; nt < 4; ++nt) {
      v16bf b0, b1;
      {
        const __bf16* p = &Ks[(nt * 16 + m) * KSTR];
        #pragma unroll
        for (int j = 0; j < 8; ++j) {
          int kk = 16 * h + 2 * j;
          b0[2 * j]     = p[kk];      b0[2 * j + 1] = p[kk + 1];
          b1[2 * j]     = p[kk + 32]; b1[2 * j + 1] = p[kk + 33];
        }
      }
      v8f acc[2] = {};
      acc[0] = __builtin_amdgcn_wmma_f32_16x16x32_bf16(false, aQ[0][0], false, b0, (short)0, acc[0], false, false);
      acc[0] = __builtin_amdgcn_wmma_f32_16x16x32_bf16(false, aQ[0][1], false, b1, (short)0, acc[0], false, false);
      acc[1] = __builtin_amdgcn_wmma_f32_16x16x32_bf16(false, aQ[1][0], false, b0, (short)0, acc[1], false, false);
      acc[1] = __builtin_amdgcn_wmma_f32_16x16x32_bf16(false, aQ[1][1], false, b1, (short)0, acc[1], false, false);

      #pragma unroll
      for (int t2 = 0; t2 < 2; ++t2) {
        #pragma unroll
        for (int r = 0; r < 8; ++r) {
          int i = t2 * 8 + r;
          float wv = EXP2F(acc[t2][r] * scl - Mrow[i]) * invS[i];
          int row  = wid * 32 + t2 * 16 + r + 8 * h;   // C/D layout row in tile
          int gcol = c * CH + nt * 16 + m;
          Wg[(size_t)row * SEQ + gcol] = wv;           // f32 attention weight (written once)
          QW[row * QSTR + nt * 16 + m] = (__bf16)wv;   // stage for PV A-fragment
        }
      }
    }
    __syncthreads();   // LDS ordering before re-reading staged W as A-layout

    v16bf aw[2][2];
    #pragma unroll
    for (int t2 = 0; t2 < 2; ++t2) {
      const __bf16* p = &QW[(wid * 32 + t2 * 16 + m) * QSTR];
      #pragma unroll
      for (int j = 0; j < 8; ++j) {
        int kk = 16 * h + 2 * j;
        aw[t2][0][2 * j]     = p[kk];
        aw[t2][0][2 * j + 1] = p[kk + 1];
        aw[t2][1][2 * j]     = p[kk + 32];
        aw[t2][1][2 * j + 1] = p[kk + 33];
      }
    }

    #pragma unroll
    for (int dt = 0; dt < 4; ++dt) {
      v16bf b0, b1;   // B = V chunk, pair-interleaved: (k,k+1) contiguous 4B
      #pragma unroll
      for (int j = 0; j < 8; ++j) {
        int dc = (dt * 16 + m) * 2;
        b0[2 * j]     = Vp[(8 * h + j) * VPSTR + dc];
        b0[2 * j + 1] = Vp[(8 * h + j) * VPSTR + dc + 1];
        b1[2 * j]     = Vp[(8 * h + j + 16) * VPSTR + dc];
        b1[2 * j + 1] = Vp[(8 * h + j + 16) * VPSTR + dc + 1];
      }
      oacc[0][dt] = __builtin_amdgcn_wmma_f32_16x16x32_bf16(false, aw[0][0], false, b0, (short)0, oacc[0][dt], false, false);
      oacc[0][dt] = __builtin_amdgcn_wmma_f32_16x16x32_bf16(false, aw[0][1], false, b1, (short)0, oacc[0][dt], false, false);
      oacc[1][dt] = __builtin_amdgcn_wmma_f32_16x16x32_bf16(false, aw[1][0], false, b0, (short)0, oacc[1][dt], false, false);
      oacc[1][dt] = __builtin_amdgcn_wmma_f32_16x16x32_bf16(false, aw[1][1], false, b1, (short)0, oacc[1][dt], false, false);
    }
  }

  // ---------------- store output tile ----------------
  #pragma unroll
  for (int t2 = 0; t2 < 2; ++t2)
    #pragma unroll
    for (int dt = 0; dt < 4; ++dt)
      #pragma unroll
      for (int r = 0; r < 8; ++r) {
        int row = wid * 32 + t2 * 16 + r + 8 * h;
        Og[(size_t)row * DHEAD + dt * 16 + m] = oacc[t2][dt][r];
      }
}

extern "C" void kernel_launch(void* const* d_in, const int* in_sizes, int n_in,
                              void* d_out, int out_size, void* d_ws, size_t ws_size,
                              hipStream_t stream) {
  (void)in_sizes; (void)n_in; (void)out_size; (void)d_ws; (void)ws_size;
  const float* q = (const float*)d_in[0];
  const float* k = (const float*)d_in[1];
  const float* v = (const float*)d_in[2];
  float* out = (float*)d_out;
  float* wgt = out + (size_t)BATCH * HEADS * SEQ * DHEAD;  // outputs concatenated: (O, W)
  dim3 grid(SEQ / TQ, BATCH * HEADS);
  attn_fwd<<<grid, 256, 0, stream>>>(q, k, v, out, wgt);
}
